// ChildSumTreeLSTM_53223234732109
// MI455X (gfx1250) — matrix operand order repaired and checked by reading, not compiled, not run on hardware
//
#include <hip/hip_runtime.h>
#include <hip/hip_bf16.h>

// ---------------------------------------------------------------------------
// Child-Sum TreeLSTM on MI455X (gfx1250)
//   N=2048 nodes, D=H=1024, post-order tree (parent[t] in [t+1, t+8])
//
// Phase A (parallel):  gx[N][4H] = x @ Wx^T + bx  via v_wmma_f32_16x16x32_bf16
// Phase B (sequential):tree scan, Wh resident in LDS as packed bf16,
//                      device-wide sense-reversing barriers per step.
// ---------------------------------------------------------------------------

#define TN 2048
#define TD 1024
#define TH 1024
#define FOURH 4096
#define SCAN_BLOCKS 64          // each owns 16 h-indices x 4 gates = 64 Wh rows

typedef __attribute__((ext_vector_type(16))) __bf16 v16bf;
typedef __attribute__((ext_vector_type(8)))  float  v8f;

union Frag32B { uint4 q[2]; v16bf v; };

__device__ __forceinline__ unsigned short f32_to_bf16_rne(float x) {
    unsigned b = __float_as_uint(x);
    b += 0x7FFFu + ((b >> 16) & 1u);
    return (unsigned short)(b >> 16);
}
__device__ __forceinline__ float sigmoidf_(float x) { return 1.0f / (1.0f + expf(-x)); }

// ------------------------------- init --------------------------------------
__global__ void k_init(float* acc, size_t n_floats, unsigned* bar) {
    size_t i = (size_t)blockIdx.x * blockDim.x + threadIdx.x;
    size_t stride = (size_t)gridDim.x * blockDim.x;
    for (size_t k = i; k < n_floats; k += stride) acc[k] = 0.0f;
    if (blockIdx.x == 0 && threadIdx.x == 0) { bar[0] = 0u; bar[1] = 0u; }
}

// --------------------------- gather + cast ---------------------------------
__global__ void k_gather(const int* __restrict__ inputs,
                         const float* __restrict__ emb,
                         unsigned short* __restrict__ xb) {
    int n = blockIdx.x;
    int row = inputs[n];
    const float* src = emb + (size_t)row * TD;
    unsigned short* dst = xb + (size_t)n * TD;
    for (int d = threadIdx.x; d < TD; d += blockDim.x)
        dst[d] = f32_to_bf16_rne(src[d]);
}

__global__ void k_castwx(const float* __restrict__ wx,
                         unsigned short* __restrict__ wxb, size_t n_elems) {
    size_t i = (size_t)blockIdx.x * blockDim.x + threadIdx.x;
    size_t stride = (size_t)gridDim.x * blockDim.x;
    for (size_t k = i; k < n_elems; k += stride) wxb[k] = f32_to_bf16_rne(wx[k]);
}

// ------------------------------- GEMM --------------------------------------
// gx[m][col] = sum_k x[m][k] * Wx_flat[col][k] + bx[col]
// A tile 16x32 bf16, B tile 32x16 bf16 (B columns = Wx rows, so both fragments
// load consecutive-K bytes). One wave -> 16(M) x 64(N) strip (A reused 4x).
__global__ __launch_bounds__(256) void k_gemm(const unsigned short* __restrict__ xb,
                                              const unsigned short* __restrict__ wxb,
                                              const float* __restrict__ bx,
                                              float* __restrict__ gx) {
    const int wave = threadIdx.x >> 5;
    const int lane = threadIdx.x & 31;
    const int task = blockIdx.x * 8 + wave;     // 1024 blocks * 8 waves = 8192
    const int mtile = task & 127;               // 128 M tiles
    const int nstrip = task >> 7;               // 64 N strips of 64
    const int m0 = mtile * 16;
    const int n0 = nstrip * 64;

    const int l15   = lane & 15;
    const int half  = lane >> 4;                // 0: lanes 0-15, 1: lanes 16-31
    const int koffA = half * 16;                // A: 16-wide K chunks
    const int koffB = half * 16;                // B: 16-wide K chunks

    const unsigned short* arow = xb + (size_t)(m0 + l15) * TD;
    const unsigned short* b0 = wxb + (size_t)(n0 +  0 + l15) * TD;
    const unsigned short* b1 = wxb + (size_t)(n0 + 16 + l15) * TD;
    const unsigned short* b2 = wxb + (size_t)(n0 + 32 + l15) * TD;
    const unsigned short* b3 = wxb + (size_t)(n0 + 48 + l15) * TD;

    v8f c0 = {}, c1 = {}, c2 = {}, c3 = {};

    for (int k = 0; k < TD; k += 32) {
        Frag32B a, f0, f1, f2, f3;
        a.q[0]  = *(const uint4*)(arow + k + koffA);
        a.q[1]  = *(const uint4*)(arow + k + koffA + 8);
        f0.q[0] = *(const uint4*)(b0 + k + koffB);
        f0.q[1] = *(const uint4*)(b0 + k + koffB + 8);
        f1.q[0] = *(const uint4*)(b1 + k + koffB);
        f1.q[1] = *(const uint4*)(b1 + k + koffB + 8);
        f2.q[0] = *(const uint4*)(b2 + k + koffB);
        f2.q[1] = *(const uint4*)(b2 + k + koffB + 8);
        f3.q[0] = *(const uint4*)(b3 + k + koffB);
        f3.q[1] = *(const uint4*)(b3 + k + koffB + 8);
        c0 = __builtin_amdgcn_wmma_f32_16x16x32_bf16(false, a.v, false, f0.v, (short)0, c0, false, false);
        c1 = __builtin_amdgcn_wmma_f32_16x16x32_bf16(false, a.v, false, f1.v, (short)0, c1, false, false);
        c2 = __builtin_amdgcn_wmma_f32_16x16x32_bf16(false, a.v, false, f2.v, (short)0, c2, false, false);
        c3 = __builtin_amdgcn_wmma_f32_16x16x32_bf16(false, a.v, false, f3.v, (short)0, c3, false, false);
    }

    // C layout: VGPR r -> M = m0 + r + 8*half, N = lane&15
    const int rbase = m0 + (half << 3);
    #pragma unroll
    for (int j = 0; j < 4; ++j) {
        int col = n0 + j * 16 + l15;
        float bias = bx[col];
        v8f c = (j == 0) ? c0 : (j == 1) ? c1 : (j == 2) ? c2 : c3;
        #pragma unroll
        for (int r = 0; r < 8; ++r)
            gx[(size_t)(rbase + r) * FOURH + col] = c[r] + bias;
    }
}

// --------------------------- device-wide barrier ---------------------------
__device__ __forceinline__ void grid_barrier(unsigned* cnt, unsigned* gen, unsigned nb) {
    __syncthreads();
    if (threadIdx.x == 0) {
        __threadfence();
        unsigned old = __hip_atomic_load(gen, __ATOMIC_RELAXED, __HIP_MEMORY_SCOPE_AGENT);
        unsigned prev = __hip_atomic_fetch_add(cnt, 1u, __ATOMIC_ACQ_REL, __HIP_MEMORY_SCOPE_AGENT);
        if (prev == nb - 1u) {
            __hip_atomic_store(cnt, 0u, __ATOMIC_RELAXED, __HIP_MEMORY_SCOPE_AGENT);
            __hip_atomic_fetch_add(gen, 1u, __ATOMIC_ACQ_REL, __HIP_MEMORY_SCOPE_AGENT);
        } else {
            while (__hip_atomic_load(gen, __ATOMIC_ACQUIRE, __HIP_MEMORY_SCOPE_AGENT) == old)
                __builtin_amdgcn_s_sleep(1);
        }
        __threadfence();
    }
    __syncthreads();
}

// ------------------------------- scan --------------------------------------
// 64 blocks x 256 threads. Block b owns h-indices [16b, 16b+16) for all 4
// gates: 64 Wh rows, kept in LDS as packed bf16 pairs (128 KB).
__global__ __launch_bounds__(256) void k_scan(const int* __restrict__ parent,
                                              const float* __restrict__ Wh,
                                              const float* __restrict__ bh,
                                              const float* __restrict__ gx,
                                              float* __restrict__ acc_h,
                                              float* __restrict__ acc_fc,
                                              unsigned* __restrict__ bar,
                                              float* __restrict__ out) {
    __shared__ unsigned WhS[64][512];   // 64 rows x 1024 bf16 (packed pairs) = 128 KB
    __shared__ float vecS[TH];          // staged hsum (phase1) / full h (phase2)
    __shared__ float gbuf[48];          // i,o,u pre-activations for 16 h-indices
    __shared__ float cbuf[16];          // c values carried phase1 -> phase2

    const int tid = threadIdx.x;
    const int blk = blockIdx.x;
    unsigned* bcnt = bar;
    unsigned* bgen = bar + 1;

    // Preload this block's Wh slice -> LDS (bf16-packed). Row lr: gate=lr/16,
    // habs = 16*blk + (lr&15).
    for (int idx = tid; idx < 64 * 512; idx += 256) {
        int lr = idx >> 9;
        int kk = idx & 511;
        int gate = lr >> 4;
        int habs = blk * 16 + (lr & 15);
        const float* src = Wh + ((size_t)(gate * TH + habs)) * TH + 2 * kk;
        unsigned a = (unsigned)f32_to_bf16_rne(src[0]);
        unsigned b = (unsigned)f32_to_bf16_rne(src[1]);
        WhS[lr][kk] = a | (b << 16);
    }
    __syncthreads();

    const int lr  = tid >> 2;   // 0..63
    const int seg = tid & 3;    // K segment: 256 floats / 128 packed pairs

    for (int t = 0; t < TN; ++t) {
        grid_barrier(bcnt, bgen, SCAN_BLOCKS);   // prior step's phase2 visible

        // ---- phase 1: gates i,o,u from hsum = acc_h[t] ----
        for (int k = tid; k < TH; k += 256) vecS[k] = acc_h[(size_t)t * TH + k];
        __syncthreads();

        float s = 0.0f;
        if (lr < 48) {
            const unsigned* wp = &WhS[lr][seg * 128];
            const float* hv = &vecS[seg * 256];
            #pragma unroll 8
            for (int kk = 0; kk < 128; ++kk) {
                unsigned p = wp[kk];
                s = fmaf(__uint_as_float(p << 16),        hv[2 * kk],     s);
                s = fmaf(__uint_as_float(p & 0xFFFF0000u), hv[2 * kk + 1], s);
            }
        }
        s += __shfl_xor(s, 1, 32);
        s += __shfl_xor(s, 2, 32);
        if (lr < 48 && seg == 0) {
            int gate = lr >> 4;
            int habs = blk * 16 + (lr & 15);
            gbuf[lr] = gx[(size_t)t * FOURH + gate * TH + habs] + bh[gate * TH + habs] + s;
        }
        __syncthreads();

        if (tid < 16) {
            int habs = blk * 16 + tid;
            float iv = sigmoidf_(gbuf[tid]);
            float ov = sigmoidf_(gbuf[16 + tid]);
            float uv = tanhf(gbuf[32 + tid]);
            float c  = iv * uv + acc_fc[(size_t)t * TH + habs];
            float h  = ov * tanhf(c);
            cbuf[tid] = c;
            out[(size_t)t * TH + habs] = h;
            if (t == TN - 1) {                    // root: emit (c, h) final state
                out[(size_t)TN * TH + habs]      = c;
                out[(size_t)TN * TH + TH + habs] = h;
            }
        }
        __threadfence();
        grid_barrier(bcnt, bgen, SCAN_BLOCKS);   // full h[t] now globally visible

        // ---- phase 2: f = sigma(fx(parent) + Wh_f @ h + bh_f); accumulate ----
        for (int k = tid; k < TH; k += 256) vecS[k] = out[(size_t)t * TH + k];
        __syncthreads();

        float s2 = 0.0f;
        if (tid < 64) {
            int lr2 = 48 + (tid >> 2);
            const unsigned* wp = &WhS[lr2][(tid & 3) * 128];
            const float* hv = &vecS[(tid & 3) * 256];
            #pragma unroll 8
            for (int kk = 0; kk < 128; ++kk) {
                unsigned p = wp[kk];
                s2 = fmaf(__uint_as_float(p << 16),        hv[2 * kk],     s2);
                s2 = fmaf(__uint_as_float(p & 0xFFFF0000u), hv[2 * kk + 1], s2);
            }
        }
        s2 += __shfl_xor(s2, 1, 32);
        s2 += __shfl_xor(s2, 2, 32);
        if (tid < 64 && (tid & 3) == 0) {
            int hidx = tid >> 2;
            int habs = blk * 16 + hidx;
            int p = parent[t];
            float fpre = bh[3 * TH + habs] + s2;
            if (p < TN) fpre += gx[(size_t)p * FOURH + 3 * TH + habs];
            float f = sigmoidf_(fpre);
            float c = cbuf[hidx];
            float h = vecS[habs];
            acc_h [(size_t)p * TH + habs] += h;        // only this block touches habs
            acc_fc[(size_t)p * TH + habs] += f * c;
        }
        __threadfence();
        // next iteration's barrier orders these writes before the next read
    }
}

// ------------------------------- launch ------------------------------------
extern "C" void kernel_launch(void* const* d_in, const int* in_sizes, int n_in,
                              void* d_out, int out_size, void* d_ws, size_t ws_size,
                              hipStream_t stream) {
    const int*   inputs = (const int*)  d_in[0];
    const int*   parent = (const int*)  d_in[1];
    const float* emb    = (const float*)d_in[2];
    const float* Wx     = (const float*)d_in[3];
    const float* bx     = (const float*)d_in[4];
    const float* Wh     = (const float*)d_in[5];
    const float* bh     = (const float*)d_in[6];
    float* out = (float*)d_out;

    // workspace layout (bytes)
    char* ws = (char*)d_ws;
    const size_t SZ_ACC  = (size_t)(TN + 1) * TH * sizeof(float);   // 8,392,704
    float*          acc_h  = (float*)(ws);
    float*          acc_fc = (float*)(ws + SZ_ACC);
    float*          gx     = (float*)(ws + 2 * SZ_ACC);
    unsigned short* xb     = (unsigned short*)(ws + 2 * SZ_ACC + (size_t)TN * FOURH * 4);
    unsigned short* wxb    = (unsigned short*)((char*)xb + (size_t)TN * TD * 2);
    unsigned*       bar    = (unsigned*)((char*)wxb + (size_t)FOURH * TD * 2);

    k_init<<<2048, 256, 0, stream>>>(acc_h, 2 * (size_t)(TN + 1) * TH, bar);
    k_gather<<<TN, 256, 0, stream>>>(inputs, emb, xb);
    k_castwx<<<2048, 256, 0, stream>>>(Wx, wxb, (size_t)FOURH * TD);
    k_gemm<<<1024, 256, 0, stream>>>(xb, wxb, bx, gx);
    k_scan<<<SCAN_BLOCKS, 256, 0, stream>>>(parent, Wh, bh, gx, acc_h, acc_fc, bar, out);
}
